// NeuroDegGraphLocalization_13211319402653
// MI455X (gfx1250) — matrix operand order repaired and checked by reading, not compiled, not run on hardware
//
#include <hip/hip_runtime.h>

#define NN   20000
#define TT   512
#define HIDC 512
#define LC   256
#define EE   320000

typedef __attribute__((ext_vector_type(16))) __bf16 v16bf;
typedef __attribute__((ext_vector_type(8)))  float  v8f;

// ---------- helpers ----------
__device__ __forceinline__ unsigned short f2bf(float f) {
    unsigned u = __float_as_uint(f);
    u += 0x7fffu + ((u >> 16) & 1u);          // round-to-nearest-even
    return (unsigned short)(u >> 16);
}
__device__ __forceinline__ float bf2f(unsigned short h) {
    return __uint_as_float(((unsigned)h) << 16);
}
__device__ __forceinline__ unsigned fkey(float f) {   // monotone float->uint
    unsigned u = __float_as_uint(f);
    return (u & 0x80000000u) ? ~u : (u | 0x80000000u);
}
__device__ __forceinline__ float kdec(unsigned k) {
    return __uint_as_float((k & 0x80000000u) ? (k ^ 0x80000000u) : ~k);
}

// ---------- generic utility kernels ----------
__global__ void zero_u32(unsigned* __restrict__ p, long long n) {
    long long i = blockIdx.x * (long long)blockDim.x + threadIdx.x;
    if (i < n) p[i] = 0u;
}

__global__ void transpose_bf16(const float* __restrict__ W, unsigned short* __restrict__ Wt,
                               int K, int Nc) {
    long long i = blockIdx.x * (long long)blockDim.x + threadIdx.x;
    if (i >= (long long)K * Nc) return;
    int k = (int)(i / Nc), n = (int)(i % Nc);
    Wt[(size_t)n * K + k] = f2bf(W[i]);
}

// ---------- GIN scatter (segment_sum of source rows into dst rows) ----------
__global__ void scatter_rows_f32(const float* __restrict__ X, const int* __restrict__ src,
                                 const int* __restrict__ dst, float* __restrict__ agg,
                                 long long total) {   // total = E*512
    long long i = blockIdx.x * (long long)blockDim.x + threadIdx.x;
    if (i >= total) return;
    int e = (int)(i >> 9), f = (int)(i & 511);
    atomicAdd(&agg[(size_t)dst[e] * 512 + f], X[(size_t)src[e] * 512 + f]);
}
__global__ void scatter_rows_bf16(const unsigned short* __restrict__ X, const int* __restrict__ src,
                                  const int* __restrict__ dst, float* __restrict__ agg,
                                  long long total) {
    long long i = blockIdx.x * (long long)blockDim.x + threadIdx.x;
    if (i >= total) return;
    int e = (int)(i >> 9), f = (int)(i & 511);
    atomicAdd(&agg[(size_t)dst[e] * 512 + f], bf2f(X[(size_t)src[e] * 512 + f]));
}
__global__ void add_ff_bf16(const float* __restrict__ a, const float* __restrict__ b,
                            unsigned short* __restrict__ o, long long n) {
    long long i = blockIdx.x * (long long)blockDim.x + threadIdx.x;
    if (i < n) o[i] = f2bf(a[i] + b[i]);
}
__global__ void add_bf_bf16(const unsigned short* __restrict__ a, const float* __restrict__ b,
                            unsigned short* __restrict__ o, long long n) {
    long long i = blockIdx.x * (long long)blockDim.x + threadIdx.x;
    if (i < n) o[i] = f2bf(bf2f(a[i]) + b[i]);
}

// ---------- WMMA GEMM: C[M,Nc] = A[M,K](bf16) * Wt[Nc,K]^T(bf16) + bias ----------
// one wave -> one 16x16 C tile; relies on 192MB L2 for A/W tile reuse.
template<int RELU, int OUT_BF16>
__global__ __launch_bounds__(256) void wmma_gemm_kernel(
        const unsigned short* __restrict__ A, const unsigned short* __restrict__ Wt,
        const float* __restrict__ bias, void* __restrict__ Cout, int M, int K, int Nc) {
    int lane = threadIdx.x;                 // 0..31
    int tile_m = blockIdx.y * blockDim.y + threadIdx.y;
    int tile_n = blockIdx.x;
    int mtiles = M >> 4;
    if (tile_m >= mtiles) return;           // wave-uniform; EXEC stays all-ones
    int hl = lane >> 4, l15 = lane & 15;

    const unsigned short* arow = A  + (size_t)(tile_m * 16 + l15) * K;
    const unsigned short* brow = Wt + (size_t)(tile_n * 16 + l15) * K + hl * 16;

    v8f acc = {};
    for (int kk = 0; kk < K; kk += 32) {
        union { v16bf v; uint4 q[2]; } a, b;
        a.q[0] = *(const uint4*)(arow + kk + hl * 16);       // K = kk+hl*16 .. +7
        a.q[1] = *(const uint4*)(arow + kk + hl * 16 + 8);   // K = kk+hl*16+8 .. +15
        const uint4* bp = (const uint4*)(brow + kk);         // K = kk+hl*16 .. +15
        b.q[0] = bp[0];
        b.q[1] = bp[1];
        acc = __builtin_amdgcn_wmma_f32_16x16x32_bf16(false, a.v, false, b.v,
                                                      (short)0, acc, false, false);
    }
    int col = tile_n * 16 + l15;
    float bs = bias ? bias[col] : 0.f;
#pragma unroll
    for (int v = 0; v < 8; ++v) {
        int row = tile_m * 16 + v + hl * 8;
        float val = acc[v] + bs;
        if (RELU) val = val > 0.f ? val : 0.f;
        if (OUT_BF16) ((unsigned short*)Cout)[(size_t)row * Nc + col] = f2bf(val);
        else          ((float*)Cout)[(size_t)row * Nc + col] = val;
    }
}

// ---------- BatchNorm over node batch ----------
__global__ void bn_stats(const float* __restrict__ h, float* __restrict__ mu,
                         float* __restrict__ rs) {
    __shared__ float sh[256];
    int c = blockIdx.x;
    float s = 0.f, ss = 0.f;
    for (int r = threadIdx.x; r < NN; r += 256) {
        float v = h[(size_t)r * LC + c]; s += v; ss += v * v;
    }
    sh[threadIdx.x] = s; __syncthreads();
    for (int k = 128; k > 0; k >>= 1) { if (threadIdx.x < k) sh[threadIdx.x] += sh[threadIdx.x + k]; __syncthreads(); }
    float sumv = sh[0]; __syncthreads();
    sh[threadIdx.x] = ss; __syncthreads();
    for (int k = 128; k > 0; k >>= 1) { if (threadIdx.x < k) sh[threadIdx.x] += sh[threadIdx.x + k]; __syncthreads(); }
    if (threadIdx.x == 0) {
        float m = sumv / NN;
        float var = sh[0] / NN - m * m;       // biased variance (training stats)
        mu[c] = m; rs[c] = rsqrtf(var + 1e-5f);
    }
}
__global__ void bn_apply(const float* __restrict__ h, const float* __restrict__ mu,
                         const float* __restrict__ rs, const float* __restrict__ g,
                         const float* __restrict__ b, unsigned short* __restrict__ o,
                         long long n) {
    long long i = blockIdx.x * (long long)blockDim.x + threadIdx.x;
    if (i >= n) return;
    int c = (int)(i & (LC - 1));
    o[i] = f2bf((h[i] - mu[c]) * rs[c] * g[c] + b[c]);
}

// ---------- GATv2 attention ----------
__global__ void edge_scores(const float* __restrict__ xl, const float* __restrict__ xr,
                            const int* __restrict__ src, const int* __restrict__ dst,
                            const float* __restrict__ att, float* __restrict__ ebuf,
                            unsigned* __restrict__ mkey) {
    long long i = blockIdx.x * (long long)blockDim.x + threadIdx.x;
    if (i >= (long long)EE * 4) return;
    int e = (int)(i >> 2), h = (int)(i & 3);
    const float* pl = xl + (size_t)src[e] * LC + h * 64;
    const float* pr = xr + (size_t)dst[e] * LC + h * 64;
    const float* pa = att + h * 64;
    float s = 0.f;
    for (int d = 0; d < 64; ++d) {
        float v = pl[d] + pr[d];
        v = v > 0.f ? v : 0.2f * v;           // leaky_relu(0.2)
        s += pa[d] * v;
    }
    ebuf[(size_t)e * 4 + h] = s;
    atomicMax(&mkey[(size_t)dst[e] * 4 + h], fkey(s));   // segment_max
}
__global__ void edge_exp(float* __restrict__ ebuf, const int* __restrict__ dst,
                         const unsigned* __restrict__ mkey, float* __restrict__ denom) {
    long long i = blockIdx.x * (long long)blockDim.x + threadIdx.x;
    if (i >= (long long)EE * 4) return;
    int e = (int)(i >> 2), h = (int)(i & 3);
    float t = __expf(ebuf[i] - kdec(mkey[(size_t)dst[e] * 4 + h]));
    ebuf[i] = t;
    atomicAdd(&denom[(size_t)dst[e] * 4 + h], t);
}
__global__ void edge_alpha(float* __restrict__ ebuf, const int* __restrict__ dst,
                           const float* __restrict__ denom) {
    long long i = blockIdx.x * (long long)blockDim.x + threadIdx.x;
    if (i >= (long long)EE * 4) return;
    int e = (int)(i >> 2), h = (int)(i & 3);
    ebuf[i] = ebuf[i] / denom[(size_t)dst[e] * 4 + h];
}
__global__ void gat_scatter(const float* __restrict__ xl, const float* __restrict__ alpha,
                            const int* __restrict__ src, const int* __restrict__ dst,
                            float* __restrict__ gout, long long total) {   // E*256
    long long i = blockIdx.x * (long long)blockDim.x + threadIdx.x;
    if (i >= total) return;
    int e = (int)(i >> 8), f = (int)(i & 255), h = f >> 6;
    atomicAdd(&gout[(size_t)dst[e] * LC + f],
              alpha[(size_t)e * 4 + h] * xl[(size_t)src[e] * LC + f]);
}
__global__ void gat_bias(float* __restrict__ g, const float* __restrict__ b, long long n) {
    long long i = blockIdx.x * (long long)blockDim.x + threadIdx.x;
    if (i < n) g[i] += b[i & (LC - 1)];
}

// ---------- LayerNorm(N*L) + Linear + sigmoid ----------
__global__ void reduce_sum_sq(const float* __restrict__ x, long long n, float* __restrict__ red) {
    __shared__ float sh[256];
    long long stride = (long long)gridDim.x * blockDim.x;
    float s = 0.f, ss = 0.f;
    for (long long i = blockIdx.x * (long long)blockDim.x + threadIdx.x; i < n; i += stride) {
        float v = x[i]; s += v; ss += v * v;
    }
    sh[threadIdx.x] = s; __syncthreads();
    for (int k = 128; k > 0; k >>= 1) { if (threadIdx.x < k) sh[threadIdx.x] += sh[threadIdx.x + k]; __syncthreads(); }
    if (threadIdx.x == 0) atomicAdd(&red[0], sh[0]);
    __syncthreads();
    sh[threadIdx.x] = ss; __syncthreads();
    for (int k = 128; k > 0; k >>= 1) { if (threadIdx.x < k) sh[threadIdx.x] += sh[threadIdx.x + k]; __syncthreads(); }
    if (threadIdx.x == 0) atomicAdd(&red[1], sh[0]);
}
__global__ void ln_dot(const float* __restrict__ x, const float* __restrict__ g,
                       const float* __restrict__ b, const float* __restrict__ Wp,
                       const float* __restrict__ red, float* __restrict__ acc, long long n) {
    __shared__ float sh[256];
    float mu = red[0] / (float)n;
    float var = red[1] / (float)n - mu * mu;
    float rs = rsqrtf(var + 1e-5f);
    long long stride = (long long)gridDim.x * blockDim.x;
    float p = 0.f;
    for (long long i = blockIdx.x * (long long)blockDim.x + threadIdx.x; i < n; i += stride)
        p += ((x[i] - mu) * rs * g[i] + b[i]) * Wp[i];
    sh[threadIdx.x] = p; __syncthreads();
    for (int k = 128; k > 0; k >>= 1) { if (threadIdx.x < k) sh[threadIdx.x] += sh[threadIdx.x + k]; __syncthreads(); }
    if (threadIdx.x == 0) atomicAdd(acc, sh[0]);
}
__global__ void finalize_pred(const float* __restrict__ acc, const float* __restrict__ bp,
                              float* __restrict__ out) {
    out[0] = 1.f / (1.f + __expf(-(acc[0] + bp[0])));
}

// ---------- node attention outputs ----------
__global__ void na_scatter(const float* __restrict__ ebuf /*flat 4E, view [H,E]*/,
                           const int* __restrict__ src, const int* __restrict__ dst,
                           float* __restrict__ na) {
    int i = blockIdx.x * blockDim.x + threadIdx.x;
    if (i >= EE * 4) return;
    int h = i / EE, e = i % EE;
    float v = ebuf[i];                                    // edge_w[h][e] (torch .view)
    atomicAdd(&na[(size_t)h * NN + src[e]], v);
    atomicAdd(&na[(size_t)h * NN + dst[e]], v);
}
__global__ void deg_count(const int* __restrict__ eidx, float* __restrict__ deg) {
    int i = blockIdx.x * blockDim.x + threadIdx.x;
    if (i < 2 * EE) atomicAdd(&deg[eidx[i]], 1.f);
}
__global__ void node_softmax(float* __restrict__ na, const float* __restrict__ deg) {
    __shared__ float sh[256];
    __shared__ float bm, bs;
    float* p = na + (size_t)blockIdx.x * NN;
    float mx = -3.4e38f;
    for (int n = threadIdx.x; n < NN; n += 256) {
        float t = p[n] / (deg[n] + 1e-9f);
        mx = fmaxf(mx, t);
    }
    sh[threadIdx.x] = mx; __syncthreads();
    for (int k = 128; k > 0; k >>= 1) { if (threadIdx.x < k) sh[threadIdx.x] = fmaxf(sh[threadIdx.x], sh[threadIdx.x + k]); __syncthreads(); }
    if (threadIdx.x == 0) bm = sh[0];
    __syncthreads();
    float s = 0.f;
    for (int n = threadIdx.x; n < NN; n += 256)
        s += __expf(p[n] / (deg[n] + 1e-9f) - bm);
    sh[threadIdx.x] = s; __syncthreads();
    for (int k = 128; k > 0; k >>= 1) { if (threadIdx.x < k) sh[threadIdx.x] += sh[threadIdx.x + k]; __syncthreads(); }
    if (threadIdx.x == 0) bs = sh[0];
    __syncthreads();
    for (int n = threadIdx.x; n < NN; n += 256)
        p[n] = __expf(p[n] / (deg[n] + 1e-9f) - bm) / bs;
}
__global__ void out_node(const float* __restrict__ na, float* __restrict__ out) {
    int n = blockIdx.x * blockDim.x + threadIdx.x;
    if (n >= NN) return;
    out[1 + n] = 0.25f * (na[n] + na[NN + n] + na[2 * NN + n] + na[3 * NN + n]);
}
__global__ void out_edge(const float* __restrict__ ebuf, float* __restrict__ out) {
    int e = blockIdx.x * blockDim.x + threadIdx.x;
    if (e >= EE) return;
    out[1 + NN + e] = 0.25f * (ebuf[e] + ebuf[EE + e] + ebuf[2 * EE + e] + ebuf[3 * EE + e]);
}

// ---------- host orchestration ----------
extern "C" void kernel_launch(void* const* d_in, const int* in_sizes, int n_in,
                              void* d_out, int out_size, void* d_ws, size_t ws_size,
                              hipStream_t stream) {
    (void)in_sizes; (void)n_in; (void)out_size; (void)ws_size;
    const float* x     = (const float*)d_in[0];
    const int*   eidx  = (const int*)d_in[1];
    const float* W1a   = (const float*)d_in[3];
    const float* b1a   = (const float*)d_in[4];
    const float* W1b   = (const float*)d_in[5];
    const float* b1b   = (const float*)d_in[6];
    const float* W2a   = (const float*)d_in[7];
    const float* b2a   = (const float*)d_in[8];
    const float* W2b   = (const float*)d_in[9];
    const float* b2b   = (const float*)d_in[10];
    const float* bn_g  = (const float*)d_in[11];
    const float* bn_b  = (const float*)d_in[12];
    const float* Wl    = (const float*)d_in[13];
    const float* Wr    = (const float*)d_in[14];
    const float* att   = (const float*)d_in[15];
    const float* gat_b = (const float*)d_in[16];
    const float* ln_g  = (const float*)d_in[17];
    const float* ln_b  = (const float*)d_in[18];
    const float* Wp    = (const float*)d_in[19];
    const float* bp    = (const float*)d_in[20];
    const int* src = eidx;
    const int* dst = eidx + EE;
    float* out = (float*)d_out;

    // workspace carve-out (phases reuse bufA/B/C/D; aliasing checked per phase)
    char* ws = (char*)d_ws;
    size_t off = 0;
    auto carve = [&](size_t bytes) -> char* {
        char* p = ws + off; off += (bytes + 255) & ~(size_t)255; return p;
    };
    unsigned short* W1aT = (unsigned short*)carve((size_t)TT * HIDC * 2);
    unsigned short* W1bT = (unsigned short*)carve((size_t)HIDC * HIDC * 2);
    unsigned short* W2aT = (unsigned short*)carve((size_t)HIDC * LC * 2);
    unsigned short* W2bT = (unsigned short*)carve((size_t)LC * LC * 2);
    unsigned short* WlT  = (unsigned short*)carve((size_t)LC * LC * 2);
    unsigned short* WrT  = (unsigned short*)carve((size_t)LC * LC * 2);
    float*          bufA = (float*)carve((size_t)NN * 512 * 4);          // agg / h2 / xl,xr
    unsigned short* bufB = (unsigned short*)carve((size_t)NN * 512 * 2); // z / hbn
    unsigned short* bufC = (unsigned short*)carve((size_t)NN * 512 * 2); // t1 / t2
    unsigned short* bufD = (unsigned short*)carve((size_t)NN * 512 * 2); // h1, then gat_out(f32, same bytes)
    float*    ebuf  = (float*)carve((size_t)EE * 4 * 4);
    unsigned* mkey  = (unsigned*)carve((size_t)NN * 4 * 4);
    float*    denom = (float*)carve((size_t)NN * 4 * 4);
    float*    na    = (float*)carve((size_t)NN * 4 * 4);
    float*    deg   = (float*)carve((size_t)NN * 4);
    float*    bnmu  = (float*)carve(LC * 4);
    float*    bnrs  = (float*)carve(LC * 4);
    float*    red   = (float*)carve(4 * 4);

    auto blks = [](long long n) { return dim3((unsigned)((n + 255) / 256)); };
    const long long nAgg = (long long)NN * 512;
    const long long nSc  = (long long)EE * 512;
    const long long nF2  = (long long)NN * LC;     // 5.12M
    dim3 gblk(32, 8);
    dim3 grd512(HIDC / 16, (NN / 16 + 7) / 8);
    dim3 grd256(LC / 16,  (NN / 16 + 7) / 8);

    // weight convert + transpose to bf16 [Nc,K]
    transpose_bf16<<<blks((long long)TT * HIDC), 256, 0, stream>>>(W1a, W1aT, TT, HIDC);
    transpose_bf16<<<blks((long long)HIDC * HIDC), 256, 0, stream>>>(W1b, W1bT, HIDC, HIDC);
    transpose_bf16<<<blks((long long)HIDC * LC), 256, 0, stream>>>(W2a, W2aT, HIDC, LC);
    transpose_bf16<<<blks((long long)LC * LC), 256, 0, stream>>>(W2b, W2bT, LC, LC);
    transpose_bf16<<<blks((long long)LC * LC), 256, 0, stream>>>(Wl, WlT, LC, LC);
    transpose_bf16<<<blks((long long)LC * LC), 256, 0, stream>>>(Wr, WrT, LC, LC);

    // ---- GIN layer 1 ----
    zero_u32<<<blks(nAgg), 256, 0, stream>>>((unsigned*)bufA, nAgg);
    scatter_rows_f32<<<blks(nSc), 256, 0, stream>>>(x, src, dst, bufA, nSc);
    add_ff_bf16<<<blks(nAgg), 256, 0, stream>>>(x, bufA, bufB, nAgg);          // z1 -> bf16
    wmma_gemm_kernel<1, 1><<<grd512, gblk, 0, stream>>>(bufB, W1aT, b1a, bufC, NN, TT, HIDC);
    wmma_gemm_kernel<1, 1><<<grd512, gblk, 0, stream>>>(bufC, W1bT, b1b, bufD, NN, HIDC, HIDC); // + inter-layer relu

    // ---- GIN layer 2 ----
    zero_u32<<<blks(nAgg), 256, 0, stream>>>((unsigned*)bufA, nAgg);
    scatter_rows_bf16<<<blks(nSc), 256, 0, stream>>>(bufD, src, dst, bufA, nSc);
    add_bf_bf16<<<blks(nAgg), 256, 0, stream>>>(bufD, bufA, bufB, nAgg);       // z2 -> bf16
    wmma_gemm_kernel<1, 1><<<grd256, gblk, 0, stream>>>(bufB, W2aT, b2a, bufC, NN, HIDC, LC);
    wmma_gemm_kernel<0, 0><<<grd256, gblk, 0, stream>>>(bufC, W2bT, b2b, bufA, NN, LC, LC); // h2 f32

    // ---- BatchNorm ----
    bn_stats<<<LC, 256, 0, stream>>>(bufA, bnmu, bnrs);
    bn_apply<<<blks(nF2), 256, 0, stream>>>(bufA, bnmu, bnrs, bn_g, bn_b, bufB, nF2); // hbn bf16

    // ---- GATv2 projections (overwrite h2; hbn lives in bufB) ----
    float* xl = bufA;
    float* xr = bufA + nF2;
    wmma_gemm_kernel<0, 0><<<grd256, gblk, 0, stream>>>(bufB, WlT, nullptr, xl, NN, LC, LC);
    wmma_gemm_kernel<0, 0><<<grd256, gblk, 0, stream>>>(bufB, WrT, nullptr, xr, NN, LC, LC);

    // ---- edge attention softmax ----
    zero_u32<<<blks((long long)NN * 4), 256, 0, stream>>>(mkey, (long long)NN * 4);
    zero_u32<<<blks((long long)NN * 4), 256, 0, stream>>>((unsigned*)denom, (long long)NN * 4);
    edge_scores<<<blks((long long)EE * 4), 256, 0, stream>>>(xl, xr, src, dst, att, ebuf, mkey);
    edge_exp<<<blks((long long)EE * 4), 256, 0, stream>>>(ebuf, dst, mkey, denom);
    edge_alpha<<<blks((long long)EE * 4), 256, 0, stream>>>(ebuf, dst, denom);

    // ---- weighted aggregation + bias ----
    float* gout = (float*)bufD;   // h1 dead; same byte size (NN*256*4 == NN*512*2)
    zero_u32<<<blks(nF2), 256, 0, stream>>>((unsigned*)gout, nF2);
    gat_scatter<<<blks((long long)EE * LC), 256, 0, stream>>>(xl, ebuf, src, dst, gout, (long long)EE * LC);
    gat_bias<<<blks(nF2), 256, 0, stream>>>(gout, gat_b, nF2);

    // ---- LayerNorm + Linear + sigmoid -> out[0] ----
    zero_u32<<<1, 4, 0, stream>>>((unsigned*)red, 4);
    reduce_sum_sq<<<512, 256, 0, stream>>>(gout, nF2, red);
    ln_dot<<<512, 256, 0, stream>>>(gout, ln_g, ln_b, Wp, red, red + 2, nF2);
    finalize_pred<<<1, 1, 0, stream>>>(red + 2, bp, out);

    // ---- node attention output ----
    zero_u32<<<blks((long long)NN * 4), 256, 0, stream>>>((unsigned*)na, (long long)NN * 4);
    zero_u32<<<blks((long long)NN), 256, 0, stream>>>((unsigned*)deg, (long long)NN);
    na_scatter<<<blks((long long)EE * 4), 256, 0, stream>>>(ebuf, src, dst, na);
    deg_count<<<blks((long long)EE * 2), 256, 0, stream>>>(eidx, deg);
    node_softmax<<<4, 256, 0, stream>>>(na, deg);
    out_node<<<blks((long long)NN), 256, 0, stream>>>(na, out);
    out_edge<<<blks((long long)EE), 256, 0, stream>>>(ebuf, out);
}
